// BailingMoeV2Gate_7224134992005
// MI455X (gfx1250) — matrix operand order repaired and checked by reading, not compiled, not run on hardware
//
#include <hip/hip_runtime.h>
#include <hip/hip_bf16.h>

#define TOKENS       16384
#define HIDDEN       4096
#define NUM_EXPERTS  256
#define TOP_K        8
#define N_GROUP      8
#define TOPK_GROUP   4
#define GROUP_SIZE   (NUM_EXPERTS / N_GROUP)   // 32
#define ROUTED_SCALING 2.5f

#define M_TILES 4                 // 16-row WMMA tiles per block => 64 tokens/block
#define BLK_M   (M_TILES * 16)
#define NKT     (HIDDEN / 32)     // 128 K-steps

typedef __attribute__((ext_vector_type(16))) __bf16          v16bf;
typedef __attribute__((ext_vector_type(8)))  float           v8f;
typedef __attribute__((ext_vector_type(4)))  float           f4;
typedef __attribute__((ext_vector_type(8)))  unsigned short  us8;

// ---- fp32 -> bf16 (RNE) split helpers in integer ops ----
__device__ __forceinline__ unsigned short f32_to_bf16_rne(float f) {
    unsigned int u = __float_as_uint(f);
    unsigned int r = u + 0x7FFFu + ((u >> 16) & 1u);
    return (unsigned short)(r >> 16);
}
__device__ __forceinline__ float bf16_bits_to_f32(unsigned short b) {
    return __uint_as_float(((unsigned int)b) << 16);
}

// ------------------------------------------------------------------
// Kernel 0: split gate_w [256,4096] f32 into bf16 hi/lo planes (in d_ws).
// ------------------------------------------------------------------
__global__ void split_w_kernel(const float* __restrict__ w,
                               unsigned short* __restrict__ whi,
                               unsigned short* __restrict__ wlo,
                               int n) {
    int i = blockIdx.x * blockDim.x + threadIdx.x;
    if (i < n) {
        float f = w[i];
        unsigned short h = f32_to_bf16_rne(f);
        whi[i] = h;
        wlo[i] = f32_to_bf16_rne(f - bf16_bits_to_f32(h));
    }
}

// ------------------------------------------------------------------
// Stage one 64x32 f32 x-tile as bf16 hi/lo into LDS in WMMA A-fragment
// lane order. Each thread: one row m, 8 contiguous K -> 2 global b128 in,
// 1 ds b128 out per plane. NT loads: x is streamed once, keep W hot in L2.
// ------------------------------------------------------------------
__device__ __forceinline__ void stage_tile(const float* __restrict__ x,
                                           int m0, int kt, int tid,
                                           unsigned short* __restrict__ dsthi,
                                           unsigned short* __restrict__ dstlo) {
    const int m  = tid >> 2;            // 0..63
    const int ks = (tid & 3) << 3;      // 0,8,16,24
    const float* src = x + (size_t)(m0 + m) * HIDDEN + kt * 32 + ks;
    f4 f0 = __builtin_nontemporal_load((const f4*)src);
    f4 f1 = __builtin_nontemporal_load((const f4*)(src + 4));
    float ff[8] = {f0.x, f0.y, f0.z, f0.w, f1.x, f1.y, f1.z, f1.w};
    us8 hv, lv;
    #pragma unroll
    for (int r = 0; r < 8; ++r) {
        unsigned short h = f32_to_bf16_rne(ff[r]);
        hv[r] = h;
        lv[r] = f32_to_bf16_rne(ff[r] - bf16_bits_to_f32(h));
    }
    // lane = (m%16) + 16*(ks>>4); elem base = ks&8; 8-run contiguous
    const int mt   = m >> 4;
    const int lane = (m & 15) + ((ks >> 4) << 4);
    const int off  = mt * 512 + lane * 16 + (ks & 8);
    *(us8*)(dsthi + off) = hv;
    *(us8*)(dstlo + off) = lv;
}

// ------------------------------------------------------------------
// Kernel 1: logits[T,256] = x[T,4096] * W^T via error-compensated bf16x3 WMMA.
// One block (8 waves) per 64-token strip; wave w owns experts [32w, 32w+32).
// Double-buffered LDS A staging, B fragments hoisted across 4 M-tiles.
// ------------------------------------------------------------------
__global__ __launch_bounds__(256)
void gate_gemm_kernel(const float* __restrict__ x,
                      const unsigned short* __restrict__ whi,
                      const unsigned short* __restrict__ wlo,
                      float* __restrict__ logits) {
    __shared__ __align__(32) unsigned short sAhi[2][M_TILES * 512];
    __shared__ __align__(32) unsigned short sAlo[2][M_TILES * 512];

    const int tid  = threadIdx.x;
    const int wave = tid >> 5;
    const int lane = tid & 31;
    const int half = (lane >> 4) & 1;
    const int lm   = lane & 15;

    const int m0 = blockIdx.x * BLK_M;
    const int e0 = wave * 32;

    v8f acc[M_TILES][2] = {};

    // B[k][n] = W[e0+n][k]: lane lm -> expert row e0+lm (tile0), e0+16+lm (tile1)
    const size_t brow0 = (size_t)(e0 + lm) * HIDDEN;
    const size_t brow1 = (size_t)(e0 + 16 + lm) * HIDDEN;

    stage_tile(x, m0, 0, tid, sAhi[0], sAlo[0]);
    __syncthreads();

    for (int kt = 0; kt < NKT; ++kt) {
        const int cur = kt & 1;

        // software pipeline: stage k-tile kt+1 while computing kt
        if (kt + 1 < NKT)
            stage_tile(x, m0, kt + 1, tid, sAhi[cur ^ 1], sAlo[cur ^ 1]);
        if (kt + 4 < NKT) {
            const int m = tid >> 2, ks = (tid & 3) << 3;
            __builtin_prefetch(x + (size_t)(m0 + m) * HIDDEN + (kt + 4) * 32 + ks, 0, 1);
        }

        // B fragments for this K-step: 16 contiguous bf16 (32B) per lane/tile/plane
        const size_t koff = (size_t)kt * 32 + (size_t)half * 16;
        v16bf b0h = *(const v16bf*)(whi + brow0 + koff);
        v16bf b0l = *(const v16bf*)(wlo + brow0 + koff);
        v16bf b1h = *(const v16bf*)(whi + brow1 + koff);
        v16bf b1l = *(const v16bf*)(wlo + brow1 + koff);

        #pragma unroll
        for (int mt = 0; mt < M_TILES; ++mt) {
            v16bf a_hi = *(const v16bf*)(&sAhi[cur][mt * 512 + lane * 16]);
            v16bf a_lo = *(const v16bf*)(&sAlo[cur][mt * 512 + lane * 16]);
            // D += Ahi*Bhi + Ahi*Blo + Alo*Bhi  (drop Alo*Blo: ~2^-16 rel)
            acc[mt][0] = __builtin_amdgcn_wmma_f32_16x16x32_bf16(false, a_hi, false, b0h, (short)0, acc[mt][0], false, false);
            acc[mt][0] = __builtin_amdgcn_wmma_f32_16x16x32_bf16(false, a_hi, false, b0l, (short)0, acc[mt][0], false, false);
            acc[mt][0] = __builtin_amdgcn_wmma_f32_16x16x32_bf16(false, a_lo, false, b0h, (short)0, acc[mt][0], false, false);
            acc[mt][1] = __builtin_amdgcn_wmma_f32_16x16x32_bf16(false, a_hi, false, b1h, (short)0, acc[mt][1], false, false);
            acc[mt][1] = __builtin_amdgcn_wmma_f32_16x16x32_bf16(false, a_hi, false, b1l, (short)0, acc[mt][1], false, false);
            acc[mt][1] = __builtin_amdgcn_wmma_f32_16x16x32_bf16(false, a_lo, false, b1h, (short)0, acc[mt][1], false, false);
        }
        __syncthreads();   // tile cur consumed; safe to overwrite next iteration
    }

    // Epilogue: C/D layout -> VGPR r: lanes 0-15 M=r, lanes 16-31 M=r+8; N=lane%16
    #pragma unroll
    for (int mt = 0; mt < M_TILES; ++mt) {
        #pragma unroll
        for (int r = 0; r < 8; ++r) {
            int m = m0 + mt * 16 + r + half * 8;
            logits[(size_t)m * NUM_EXPERTS + e0 + lm]      = acc[mt][0][r];
            logits[(size_t)m * NUM_EXPERTS + e0 + 16 + lm] = acc[mt][1][r];
        }
    }
}

// ------------------------------------------------------------------
// Kernel 2: group-limited top-k routing. One wave per 32 tokens, thread-per-token.
// ------------------------------------------------------------------
__global__ __launch_bounds__(32)
void route_kernel(const float* __restrict__ logits,
                  const float* __restrict__ bias,
                  float* __restrict__ out_idx,
                  float* __restrict__ out_w) {
    __shared__ float sc[32 * 257];        // 257 pitch -> no 32-way bank conflicts
    __shared__ float sb[NUM_EXPERTS];

    const int tid = threadIdx.x;
    const int t0  = blockIdx.x * 32;

    for (int e = tid; e < NUM_EXPERTS; e += 32) sb[e] = bias[e];
    for (int j = tid; j < 32 * NUM_EXPERTS; j += 32) {   // coalesced along experts
        int tl = j >> 8;
        int e  = j & 255;
        float lg = logits[(size_t)(t0 + tl) * NUM_EXPERTS + e];
        sc[tl * 257 + e] = 1.0f / (1.0f + expf(-lg));    // sigmoid score
    }
    __syncthreads();

    const float* row = &sc[tid * 257];
    const int t = t0 + tid;
    const float NEG_INF = -__builtin_inff();

    // group score = sum of top-2 (score + bias) within each group of 32
    float gs[N_GROUP];
    #pragma unroll
    for (int g = 0; g < N_GROUP; ++g) {
        float m1 = NEG_INF, m2 = NEG_INF;
        for (int j = 0; j < GROUP_SIZE; ++j) {
            float v = row[g * GROUP_SIZE + j] + sb[g * GROUP_SIZE + j];
            if (v > m1) { m2 = m1; m1 = v; }
            else if (v > m2) { m2 = v; }
        }
        gs[g] = m1 + m2;
    }

    // top-4 groups (strict '>' keeps lowest index on ties, matching lax.top_k)
    unsigned sel = 0;
    for (int it = 0; it < TOPK_GROUP; ++it) {
        float best = NEG_INF; int bi = 0;
        #pragma unroll
        for (int g = 0; g < N_GROUP; ++g) {
            if (!((sel >> g) & 1u) && gs[g] > best) { best = gs[g]; bi = g; }
        }
        sel |= (1u << bi);
    }

    // top-8 experts among selected groups
    int   chosen[TOP_K];
    float gathered[TOP_K];
    unsigned long long used[4] = {0ull, 0ull, 0ull, 0ull};
    for (int it = 0; it < TOP_K; ++it) {
        float best = NEG_INF; int bi = 0;
        for (int e = 0; e < NUM_EXPERTS; ++e) {
            if (!((sel >> (e >> 5)) & 1u)) continue;              // group masked
            if ((used[e >> 6] >> (e & 63)) & 1ull) continue;      // already taken
            float v = row[e] + sb[e];
            if (v > best) { best = v; bi = e; }
        }
        used[bi >> 6] |= (1ull << (bi & 63));
        chosen[it]   = bi;
        gathered[it] = row[bi];     // unbiased sigmoid score for the weight
    }

    float s = 1e-20f;
    #pragma unroll
    for (int i = 0; i < TOP_K; ++i) s += gathered[i];
    float scale = ROUTED_SCALING / s;

    #pragma unroll
    for (int i = 0; i < TOP_K; ++i) {
        out_idx[(size_t)t * TOP_K + i] = (float)chosen[i];  // indices 0..255 exact in f32
        out_w[(size_t)t * TOP_K + i]   = gathered[i] * scale;
    }
}

// ------------------------------------------------------------------
extern "C" void kernel_launch(void* const* d_in, const int* in_sizes, int n_in,
                              void* d_out, int out_size, void* d_ws, size_t ws_size,
                              hipStream_t stream) {
    const float* x    = (const float*)d_in[0];   // [TOKENS, HIDDEN]
    const float* gw   = (const float*)d_in[1];   // [NUM_EXPERTS, HIDDEN]
    const float* bias = (const float*)d_in[2];   // [NUM_EXPERTS]

    float* out      = (float*)d_out;
    float* out_idx  = out;                                        // [T, 8]
    float* out_w    = out + (size_t)TOKENS * TOP_K;               // [T, 8]
    float* logits   = out + (size_t)TOKENS * TOP_K * 2;           // [T, 256]

    unsigned short* whi = (unsigned short*)d_ws;                         // 2 MB
    unsigned short* wlo = whi + (size_t)NUM_EXPERTS * HIDDEN;            // 2 MB

    const int nw = NUM_EXPERTS * HIDDEN;
    split_w_kernel<<<(nw + 255) / 256, 256, 0, stream>>>(gw, whi, wlo, nw);
    gate_gemm_kernel<<<TOKENS / BLK_M, 256, 0, stream>>>(x, whi, wlo, logits);
    route_kernel<<<TOKENS / 32, 32, 0, stream>>>(logits, bias, out_idx, out_w);
}
